// BatchedTreeForest_37804302139402
// MI455X (gfx1250) — matrix operand order repaired and checked by reading, not compiled, not run on hardware
//
#include <hip/hip_runtime.h>
#include <math.h>

typedef __attribute__((ext_vector_type(8)))  float  v8f;
typedef __attribute__((ext_vector_type(8)))  __bf16 v8bf;
typedef __attribute__((ext_vector_type(16))) __bf16 v16bf;

// Problem constants (fixed by the reference)
constexpr int D   = 1024;   // feature dim
constexpr int T   = 16;     // trees
constexpr int NI  = 15;     // internal nodes per tree
constexpr int NL  = 16;     // leaves per tree
constexpr int O   = 512;    // output dim
constexpr int NC  = 256;    // combined cols: T*NI (240) decision + T (16) gate
constexpr float EPS = 1e-5f;

__device__ __forceinline__ v16bf cat8(v8bf a, v8bf b) {
    return __builtin_shufflevector(a, b, 0,1,2,3,4,5,6,7,8,9,10,11,12,13,14,15);
}

__device__ __forceinline__ v8f wmma_bf16(v16bf a, v16bf b, v8f c) {
    // D(16x16,f32) = A(16x32,bf16) * B(32x16,bf16) + C
    return __builtin_amdgcn_wmma_f32_16x16x32_bf16(
        /*neg_a=*/false, a, /*neg_b=*/false, b,
        /*c_mod=*/(short)0, c, /*reuse_a=*/false, /*reuse_b=*/false);
}

__device__ __forceinline__ void split_bf16(float v, __bf16& hi, __bf16& lo) {
    hi = (__bf16)v;
    lo = (__bf16)(v - (float)hi);
}

// ---------------------------------------------------------------------------
// Kernel 0a: pack decision weights (240xD) + gate_w^T (16xD) into bf16 hi/lo
// Wc*[n][d]: n in [0,240): n = t*15+node ; n in [240,256): gate tree t=n-240
// ---------------------------------------------------------------------------
__global__ __launch_bounds__(256)
void prep_wc(const float* __restrict__ dw, const float* __restrict__ gw,
             __bf16* __restrict__ wch, __bf16* __restrict__ wcl) {
    int n = blockIdx.x;
    for (int d = threadIdx.x; d < D; d += blockDim.x) {
        float v = (n < T * NI) ? dw[(size_t)n * D + d]
                               : gw[(size_t)d * T + (n - T * NI)];
        __bf16 hi, lo;
        split_bf16(v, hi, lo);
        wch[(size_t)n * D + d] = hi;
        wcl[(size_t)n * D + d] = lo;
    }
}

// ---------------------------------------------------------------------------
// Kernel 0b: transpose leaf_outputs (256 x 512) -> LOT*[o][k] bf16 hi/lo
// (K-contiguous per output column for WMMA B-fragment 32-byte reads)
// ---------------------------------------------------------------------------
__global__ __launch_bounds__(256)
void prep_lot(const float* __restrict__ lo,
              __bf16* __restrict__ loth, __bf16* __restrict__ lotl) {
    int o = blockIdx.x;        // 0..511
    int k = threadIdx.x;       // 0..255
    float v = lo[(size_t)k * O + o];
    __bf16 hi, ll;
    split_bf16(v, hi, ll);
    loth[(size_t)o * NC + k] = hi;
    lotl[(size_t)o * NC + k] = ll;
}

// ---------------------------------------------------------------------------
// Stage 1: logits GEMM (16 rows x 256 cols, K=1024) via split-bf16 WMMA
// (hi*hi + hi*lo + lo*hi, f32 accum), then tree/gate epilogue ->
// weighted leaf probs written as bf16 hi/lo pairs.
// Block: 128 threads = 4 waves; wave w owns cols [w*64, w*64+64).
// ---------------------------------------------------------------------------
#define XS2       536   // bf16 LDS stride for 512-K chunk (16B aligned, bank-spread)
#define LS_STRIDE 260   // f32 logits stride

__global__ __launch_bounds__(128)
void stage1(const float* __restrict__ x,
            const __bf16* __restrict__ wch, const __bf16* __restrict__ wcl,
            const float* __restrict__ dec_bias, const float* __restrict__ gate_b,
            const float* __restrict__ ntl,
            __bf16* __restrict__ wlph, __bf16* __restrict__ wlpl) {
    __shared__ __bf16 Xh[16 * XS2];
    __shared__ __bf16 Xl[16 * XS2];
    __shared__ float  Ls[16 * LS_STRIDE];

    const int rowBase = blockIdx.x * 16;
    const int tid  = threadIdx.x;
    const int wave = tid >> 5;
    const int lane = tid & 31;
    const int mrow = lane & 15;
    const int h    = lane >> 4;

    // B-fragment base pointers (K-contiguous rows), 4 column tiles per wave
    const __bf16* bph[4];
    const __bf16* bpl[4];
#pragma unroll
    for (int j = 0; j < 4; ++j) {
        int n = wave * 64 + j * 16 + (lane & 15);
        bph[j] = wch + (size_t)n * D + 16 * h;
        bpl[j] = wcl + (size_t)n * D + 16 * h;
    }

    v8f acc[4] = {v8f{}, v8f{}, v8f{}, v8f{}};

    for (int c = 0; c < 2; ++c) {          // two 512-wide K chunks
        __syncthreads();                   // previous chunk fully consumed
        // load 16 x 512 fp32 chunk, split to bf16 hi/lo in LDS
        for (int i = tid; i < 2048; i += 128) {
            int r = i >> 7, c4 = i & 127;
            float4 v = ((const float4*)(x + (size_t)(rowBase + r) * D + c * 512))[c4];
            int base = r * XS2 + c4 * 4;
            __bf16 hi, lo;
            split_bf16(v.x, hi, lo); Xh[base + 0] = hi; Xl[base + 0] = lo;
            split_bf16(v.y, hi, lo); Xh[base + 1] = hi; Xl[base + 1] = lo;
            split_bf16(v.z, hi, lo); Xh[base + 2] = hi; Xl[base + 2] = lo;
            split_bf16(v.w, hi, lo); Xh[base + 3] = hi; Xl[base + 3] = lo;
        }
        __syncthreads();

        for (int kb = 0; kb < 16; ++kb) {  // 16 k-blocks of 32 in this chunk
            const __bf16* arh = &Xh[mrow * XS2 + kb * 32];
            const __bf16* arl = &Xl[mrow * XS2 + kb * 32];
            // A frag: K run {16h..16h+15}
            v16bf Ah = cat8(*(const v8bf*)(arh + 16 * h),
                            *(const v8bf*)(arh + 16 * h + 8));
            v16bf Al = cat8(*(const v8bf*)(arl + 16 * h),
                            *(const v8bf*)(arl + 16 * h + 8));
            const int kofs = c * 512 + kb * 32;
#pragma unroll
            for (int j = 0; j < 4; ++j) {
                v16bf Bh = *(const v16bf*)(bph[j] + kofs);
                v16bf Bl = *(const v16bf*)(bpl[j] + kofs);
                acc[j] = wmma_bf16(Ah, Bh, acc[j]);
                acc[j] = wmma_bf16(Ah, Bl, acc[j]);
                acc[j] = wmma_bf16(Al, Bh, acc[j]);
            }
        }
    }

    // D-frag -> LDS logits: VGPR v holds M = v + 8h, N = lane&15
    const int mhi = 8 * h;
#pragma unroll
    for (int j = 0; j < 4; ++j) {
        int col = wave * 64 + j * 16 + (lane & 15);
#pragma unroll
        for (int v = 0; v < 8; ++v)
            Ls[(v + mhi) * LS_STRIDE + col] = acc[j][v];
    }
    __syncthreads();

    // Epilogue: 256 (row,tree) tasks over 128 threads
#pragma unroll
    for (int it = 0; it < 2; ++it) {
        int task = tid + it * 128;
        int row = task >> 4;
        int t   = task & 15;

        // gate softmax over 16 trees for this row
        float gl[16], mx = -1e30f;
#pragma unroll
        for (int u = 0; u < 16; ++u) {
            gl[u] = Ls[row * LS_STRIDE + 240 + u] + gate_b[u];
            mx = fmaxf(mx, gl[u]);
        }
        float ssum = 0.0f;
#pragma unroll
        for (int u = 0; u < 16; ++u) ssum += expf(gl[u] - mx);
        float gate = expf(gl[t] - mx) / ssum;

        // decisions with learnable softplus temperature
        float dec[15];
#pragma unroll
        for (int n = 0; n < 15; ++n) {
            float lg = Ls[row * LS_STRIDE + t * NI + n] + dec_bias[t * NI + n];
            float tl = ntl[t * NI + n] + 0.5413f;
            float temp = log1pf(expf(tl));               // TEMP * softplus
            dec[n] = 1.0f / (1.0f + expf(-lg / temp));
        }

        // 16 root->leaf path products, scaled by gate; write bf16 hi/lo
        size_t obase = (size_t)(rowBase + row) * NC + t * NL;
#pragma unroll
        for (int leaf = 0; leaf < 16; ++leaf) {
            float p = gate;
            int node = 0;
#pragma unroll
            for (int dl = 0; dl < 4; ++dl) {
                int bit = (leaf >> (3 - dl)) & 1;
                float dn = dec[node];
                p *= bit ? (1.0f - dn) : dn;
                node = 2 * node + 1 + bit;
            }
            __bf16 hi, lo;
            split_bf16(p, hi, lo);
            wlph[obase + leaf] = hi;
            wlpl[obase + leaf] = lo;
        }
    }
}

// ---------------------------------------------------------------------------
// Stage 2: out = wlp(rows x 256) * leaf_outputs(256 x 512) via split-bf16
// WMMA, fused LayerNorm over O=512 per row.
// Block: 256 threads = 8 waves; wave w owns cols [w*64, w*64+64).
// ---------------------------------------------------------------------------
#define AS2       280   // bf16 LDS stride (16B aligned, bank-spread)
#define OS_STRIDE 516   // f32 output stride

__global__ __launch_bounds__(256)
void stage2(const __bf16* __restrict__ wlph, const __bf16* __restrict__ wlpl,
            const __bf16* __restrict__ loth, const __bf16* __restrict__ lotl,
            const float* __restrict__ gamma, const float* __restrict__ beta,
            float* __restrict__ out) {
    __shared__ __bf16 Ah[16 * AS2];
    __shared__ __bf16 Al[16 * AS2];
    __shared__ float  Os[16 * OS_STRIDE];

    const int rowBase = blockIdx.x * 16;
    const int tid  = threadIdx.x;
    const int wave = tid >> 5;
    const int lane = tid & 31;
    const int mrow = lane & 15;
    const int h    = lane >> 4;

    // load A tile 16x256 bf16 hi/lo: 512 16-byte chunks each
    for (int i = tid; i < 512; i += 256) {
        int r = i >> 5, c8 = i & 31;
        *(v8bf*)&Ah[r * AS2 + c8 * 8] =
            ((const v8bf*)(wlph + (size_t)(rowBase + r) * NC))[c8];
        *(v8bf*)&Al[r * AS2 + c8 * 8] =
            ((const v8bf*)(wlpl + (size_t)(rowBase + r) * NC))[c8];
    }
    __syncthreads();

    v8f acc[4] = {v8f{}, v8f{}, v8f{}, v8f{}};

    for (int kb = 0; kb < 8; ++kb) {       // 8 k-blocks of 32
        const __bf16* arh = &Ah[mrow * AS2 + kb * 32];
        const __bf16* arl = &Al[mrow * AS2 + kb * 32];
        v16bf Afh = cat8(*(const v8bf*)(arh + 16 * h),
                         *(const v8bf*)(arh + 16 * h + 8));
        v16bf Afl = cat8(*(const v8bf*)(arl + 16 * h),
                         *(const v8bf*)(arl + 16 * h + 8));
        const int kofs = kb * 32 + 16 * h;
#pragma unroll
        for (int j = 0; j < 4; ++j) {
            int n = wave * 64 + j * 16 + (lane & 15);
            v16bf Bh = *(const v16bf*)(loth + (size_t)n * NC + kofs);
            v16bf Bl = *(const v16bf*)(lotl + (size_t)n * NC + kofs);
            acc[j] = wmma_bf16(Afh, Bh, acc[j]);
            acc[j] = wmma_bf16(Afh, Bl, acc[j]);
            acc[j] = wmma_bf16(Afl, Bh, acc[j]);
        }
    }

    const int mhi = 8 * h;
#pragma unroll
    for (int j = 0; j < 4; ++j) {
        int col = wave * 64 + j * 16 + (lane & 15);
#pragma unroll
        for (int v = 0; v < 8; ++v)
            Os[(v + mhi) * OS_STRIDE + col] = acc[j][v];
    }
    __syncthreads();

    // LayerNorm: 16 threads per row (half-wave), 32 elems per thread
    const int row = tid >> 4;
    const int l16 = tid & 15;
    float s = 0.0f, ss = 0.0f;
#pragma unroll
    for (int j = 0; j < 32; ++j) {
        float v = Os[row * OS_STRIDE + l16 + 16 * j];
        s += v; ss += v * v;
    }
#pragma unroll
    for (int m = 1; m < 16; m <<= 1) {
        s  += __shfl_xor(s,  m, 16);
        ss += __shfl_xor(ss, m, 16);
    }
    float mean = s * (1.0f / (float)O);
    float var  = ss * (1.0f / (float)O) - mean * mean;
    float inv  = rsqrtf(var + EPS);
    float* op = out + (size_t)(rowBase + row) * O;
#pragma unroll
    for (int j = 0; j < 32; ++j) {
        int o = l16 + 16 * j;
        op[o] = (Os[row * OS_STRIDE + o] - mean) * inv * gamma[o] + beta[o];
    }
}

// ---------------------------------------------------------------------------
extern "C" void kernel_launch(void* const* d_in, const int* in_sizes, int n_in,
                              void* d_out, int out_size, void* d_ws, size_t ws_size,
                              hipStream_t stream) {
    const float* x     = (const float*)d_in[0];
    const float* dw    = (const float*)d_in[1];
    const float* db    = (const float*)d_in[2];
    const float* lo    = (const float*)d_in[3];
    const float* gw    = (const float*)d_in[4];
    const float* gb    = (const float*)d_in[5];
    const float* ntl   = (const float*)d_in[6];
    const float* gamma = (const float*)d_in[7];
    const float* beta  = (const float*)d_in[8];
    float* out = (float*)d_out;

    const int rows = in_sizes[0] / D;   // B*S = 8192

    // workspace layout (bf16):
    //   wlp_hi rowsxNC | wlp_lo rowsxNC | Wc_hi NCxD | Wc_lo NCxD
    //   | LOT_hi OxNC | LOT_lo OxNC
    char* p = (char*)d_ws;
    __bf16* wlph = (__bf16*)p;                 p += (size_t)rows * NC * 2;
    __bf16* wlpl = (__bf16*)p;                 p += (size_t)rows * NC * 2;
    __bf16* wch  = (__bf16*)p;                 p += (size_t)NC * D * 2;
    __bf16* wcl  = (__bf16*)p;                 p += (size_t)NC * D * 2;
    __bf16* loth = (__bf16*)p;                 p += (size_t)O * NC * 2;
    __bf16* lotl = (__bf16*)p;

    prep_wc<<<NC, 256, 0, stream>>>(dw, gw, wch, wcl);
    prep_lot<<<O, 256, 0, stream>>>(lo, loth, lotl);
    stage1<<<rows / 16, 128, 0, stream>>>(x, wch, wcl, db, gb, ntl, wlph, wlpl);
    stage2<<<rows / 16, 256, 0, stream>>>(wlph, wlpl, loth, lotl, gamma, beta, out);
}
